// GraphSAGE_16965120819652
// MI455X (gfx1250) — matrix operand order repaired and checked by reading, not compiled, not run on hardware
//
#include <hip/hip_runtime.h>

// ---------------- problem constants ----------------
#define IN_F   128
#define HID_F  512
#define OUT_F  256

typedef __attribute__((ext_vector_type(16))) _Float16 v16h;
typedef __attribute__((ext_vector_type(8)))  _Float16 v8h;
typedef __attribute__((ext_vector_type(8)))  float    v8f;

// Packed-f16 atomic add (GLOBAL_ATOMIC_PK_ADD_F16, no return -> STOREcnt).
// data = two f16 values packed in 32 bits; addr must be 4B aligned.
__device__ __forceinline__ void pk_add_f16(_Float16* addr, unsigned int data) {
    asm volatile("global_atomic_pk_add_f16 %0, %1, off"
                 :: "v"(addr), "v"(data)
                 : "memory");
}

// ---------------------------------------------------------------------------
// Pack row-major f32 weights W[K][Ncols] into pre-swizzled f16 WMMA B-frags:
//   frag[((kt*ntiles + nt)*32 + lane)*16 + e]
//     = (f16) W[kt*32 + 8*(lane>>4) + (e&7) + 16*(e>>3)][nt*16 + (lane&15)]
// so each GEMM lane loads its 16 f16 (32B) contiguously & coalesced.
// ---------------------------------------------------------------------------
__global__ void pack_B_frag(const float* __restrict__ W, _Float16* __restrict__ frag,
                            int K, int Ncols) {
    int idx = blockIdx.x * blockDim.x + threadIdx.x;
    if (idx >= K * Ncols) return;
    int e    = idx & 15;
    int lane = (idx >> 4) & 31;
    int tile = idx >> 9;
    int ntiles = Ncols >> 4;
    int kt = tile / ntiles;
    int nt = tile - kt * ntiles;
    int k  = kt * 32 + 8 * (lane >> 4) + (e & 7) + 16 * (e >> 3);
    int n  = nt * 16 + (lane & 15);
    frag[idx] = (_Float16)W[(size_t)k * Ncols + n];
}

// f32 -> f16, 4 elements/thread
__global__ void cast_f32_to_f16x4(const float* __restrict__ src,
                                  _Float16* __restrict__ dst, int totalVec) {
    int idx = blockIdx.x * blockDim.x + threadIdx.x;
    if (idx >= totalVec) return;
    float4 v = ((const float4*)src)[idx];
    _Float16* o = dst + (size_t)idx * 4;
    o[0] = (_Float16)v.x; o[1] = (_Float16)v.y;
    o[2] = (_Float16)v.z; o[3] = (_Float16)v.w;
}

// layer-1 scatter + degree: one wave/edge; lane handles 4 of 128 f16 features.
// Two pk-f16 atomics per lane; lane 0 also bumps the degree counter.
__global__ void scatter_layer1(const long long* __restrict__ ei,
                               const _Float16* __restrict__ x16,
                               _Float16* __restrict__ agg, float* __restrict__ cnt,
                               int E) {
    int wave = __builtin_amdgcn_readfirstlane(
        blockIdx.x * (blockDim.x >> 5) + (threadIdx.x >> 5));
    if (wave >= E) return;
    int lane = threadIdx.x & 31;
    long long s = ei[wave];
    long long d = ei[(size_t)E + wave];
    uint2 xv = *(const uint2*)(x16 + (size_t)s * IN_F + lane * 4);
    _Float16* ag = agg + (size_t)d * IN_F + lane * 4;
    pk_add_f16(ag + 0, xv.x);
    pk_add_f16(ag + 2, xv.y);
    if (lane == 0) atomicAdd(&cnt[d], 1.0f);
}

// layer-2 scatter: one wave/edge; lane handles 16 of 512 f16 features.
// 2x b128 gather + 8 pk-f16 atomics per lane.
__global__ void scatter_layer2(const long long* __restrict__ ei,
                               const _Float16* __restrict__ h,
                               _Float16* __restrict__ agg, int E) {
    int wave = __builtin_amdgcn_readfirstlane(
        blockIdx.x * (blockDim.x >> 5) + (threadIdx.x >> 5));
    if (wave >= E) return;
    int lane = threadIdx.x & 31;
    long long s = ei[wave];
    long long d = ei[(size_t)E + wave];
    const uint4* hp = (const uint4*)(h + (size_t)s * HID_F + lane * 16);
    uint4 a = hp[0];
    uint4 b = hp[1];
    _Float16* ag = agg + (size_t)d * HID_F + lane * 16;
    pk_add_f16(ag + 0,  a.x); pk_add_f16(ag + 2,  a.y);
    pk_add_f16(ag + 4,  a.z); pk_add_f16(ag + 6,  a.w);
    pk_add_f16(ag + 8,  b.x); pk_add_f16(ag + 10, b.y);
    pk_add_f16(ag + 12, b.z); pk_add_f16(ag + 14, b.w);
}

// mean-normalize f16 sums -> f16 means, 8 elements/thread
__global__ void normalize_f16x8(const _Float16* __restrict__ agg,
                                const float* __restrict__ cnt,
                                _Float16* __restrict__ out, int totalVec, int log2F) {
    int idx = blockIdx.x * blockDim.x + threadIdx.x;
    if (idx >= totalVec) return;
    int node = (idx << 3) >> log2F;
    float r = 1.0f / fmaxf(cnt[node], 1.0f);
    v8h a = *(const v8h*)(agg + (size_t)idx * 8);
    v8h o;
#pragma unroll
    for (int i = 0; i < 8; ++i) o[i] = (_Float16)((float)a[i] * r);
    *(v8h*)(out + (size_t)idx * 8) = o;
}

// ---------------------------------------------------------------------------
// Fused dual GEMM: D = A0@B0 + A1@B1 + bias (optional relu, f16/f32 store).
// Register-blocked: each wave computes a 32x64 output patch
// (MT=2 m-tiles x NT=4 n-tiles, 8 f32 accumulators), so each A fragment pair
// feeds 8 v_wmma_f32_16x16x32_f16 and B fragments are reused across 2 m-tiles.
// K and NTILES are compile-time so every load in the K loop is
// base + immediate offset (24-bit IOFFSET) -- no 64-bit address chains,
// no address spills.
// ---------------------------------------------------------------------------
__device__ __forceinline__ v16h load_A_frag(const _Float16* p) {
    v8h lo = *(const v8h*)p;
    v8h hi = *(const v8h*)(p + 16);
    v16h a;
#pragma unroll
    for (int i = 0; i < 8; ++i) { a[i] = lo[i]; a[i + 8] = hi[i]; }
    return a;
}

template <bool RELU, bool STORE_F16, int K, int NTILES>
__global__ void gemm_dual_wmma(const _Float16* __restrict__ A0,
                               const _Float16* __restrict__ B0,
                               const _Float16* __restrict__ A1,
                               const _Float16* __restrict__ B1,
                               const float*    __restrict__ bias,
                               _Float16*       __restrict__ out16,
                               float*          __restrict__ out32,
                               int mgroups, int ngroups) {
    constexpr int NCOLS = NTILES * 16;
    int wave = blockIdx.x * (blockDim.x >> 5) + (threadIdx.x >> 5);
    if (wave >= mgroups * ngroups) return;        // whole-wave exit: EXEC all-1 at WMMA
    const int lane = threadIdx.x & 31;
    const int half = lane >> 4;
    const int lidx = lane & 15;
    const int mg = wave / ngroups;
    const int ng = wave - mg * ngroups;
    const int m0  = mg * 32;                      // two m-tiles: m0, m0+16
    const int nt0 = ng * 4;                       // four n-tiles

    // Per-lane base pointers; all loop offsets below are compile-time.
    const _Float16* a0p = A0 + (size_t)(m0 + lidx) * K + 8 * half;
    const _Float16* a1p = A1 + (size_t)(m0 + lidx) * K + 8 * half;
    const _Float16* b0p = B0 + ((size_t)nt0 * 32 + lane) * 16;
    const _Float16* b1p = B1 + ((size_t)nt0 * 32 + lane) * 16;

    v8f c[2][4];
#pragma unroll
    for (int mi = 0; mi < 2; ++mi)
#pragma unroll
        for (int j = 0; j < 4; ++j) c[mi][j] = (v8f){};

#pragma unroll 4
    for (int kt = 0; kt < K / 32; ++kt) {
        v16h a0 = load_A_frag(a0p + kt * 32);
        v16h a1 = load_A_frag(a0p + 16 * K + kt * 32);
#pragma unroll
        for (int j = 0; j < 4; ++j) {
            v16h b = *(const v16h*)(b0p + (size_t)kt * (NTILES * 512) + j * 512);
            c[0][j] = __builtin_amdgcn_wmma_f32_16x16x32_f16(false, a0, false, b,
                                                             (short)0, c[0][j], false, false);
            c[1][j] = __builtin_amdgcn_wmma_f32_16x16x32_f16(false, a1, false, b,
                                                             (short)0, c[1][j], false, false);
        }
    }
#pragma unroll 4
    for (int kt = 0; kt < K / 32; ++kt) {
        v16h a0 = load_A_frag(a1p + kt * 32);
        v16h a1 = load_A_frag(a1p + 16 * K + kt * 32);
#pragma unroll
        for (int j = 0; j < 4; ++j) {
            v16h b = *(const v16h*)(b1p + (size_t)kt * (NTILES * 512) + j * 512);
            c[0][j] = __builtin_amdgcn_wmma_f32_16x16x32_f16(false, a0, false, b,
                                                             (short)0, c[0][j], false, false);
            c[1][j] = __builtin_amdgcn_wmma_f32_16x16x32_f16(false, a1, false, b,
                                                             (short)0, c[1][j], false, false);
        }
    }

    // Epilogue: bias + optional relu; stores are base + immediate offsets too.
    _Float16* o16 = STORE_F16 ? out16 + (size_t)(m0 + 8 * half) * NCOLS + nt0 * 16 + lidx
                              : nullptr;
    float*    o32 = STORE_F16 ? nullptr
                              : out32 + (size_t)(m0 + 8 * half) * NCOLS + nt0 * 16 + lidx;
#pragma unroll
    for (int j = 0; j < 4; ++j) {
        const float bv = bias[(nt0 + j) * 16 + lidx];
#pragma unroll
        for (int mi = 0; mi < 2; ++mi) {
#pragma unroll
            for (int r = 0; r < 8; ++r) {
                float v = c[mi][j][r] + bv;
                if (RELU) v = fmaxf(v, 0.0f);
                const size_t ofs = (size_t)(mi * 16 + r) * NCOLS + j * 16;
                if (STORE_F16) o16[ofs] = (_Float16)v;
                else           o32[ofs] = v;
            }
        }
    }
}

// ---------------------------------------------------------------------------
extern "C" void kernel_launch(void* const* d_in, const int* in_sizes, int n_in,
                              void* d_out, int out_size, void* d_ws, size_t ws_size,
                              hipStream_t stream) {
    const float*     x   = (const float*)d_in[0];
    const long long* ei  = (const long long*)d_in[1];
    const float*     W1l = (const float*)d_in[2];
    const float*     b1  = (const float*)d_in[3];
    const float*     W1r = (const float*)d_in[4];
    const float*     W2l = (const float*)d_in[5];
    const float*     b2  = (const float*)d_in[6];
    const float*     W2r = (const float*)d_in[7];
    float* out = (float*)d_out;

    const int N = in_sizes[0] / IN_F;   // 40000
    const int E = in_sizes[1] / 2;      // 640000

    // ---- workspace layout (256B aligned slabs) ----
    char* ws = (char*)d_ws;
    size_t off = 0;
    auto alloc = [&](size_t bytes) {
        size_t o = off;
        off = (off + bytes + 255) & ~(size_t)255;
        return o;
    };
    _Float16* x16    = (_Float16*)(ws + alloc((size_t)N * IN_F  * 2));
    _Float16* a1_16  = (_Float16*)(ws + alloc((size_t)N * IN_F  * 2));
    _Float16* h16    = (_Float16*)(ws + alloc((size_t)N * HID_F * 2));
    _Float16* a2_16  = (_Float16*)(ws + alloc((size_t)N * HID_F * 2));
    _Float16* agg1   = (_Float16*)(ws + alloc((size_t)N * IN_F  * 2));
    _Float16* agg2   = (_Float16*)(ws + alloc((size_t)N * HID_F * 2));
    float*    cnt    = (float*)   (ws + alloc((size_t)N * 4));
    _Float16* fW1l   = (_Float16*)(ws + alloc((size_t)IN_F  * HID_F * 2));
    _Float16* fW1r   = (_Float16*)(ws + alloc((size_t)IN_F  * HID_F * 2));
    _Float16* fW2l   = (_Float16*)(ws + alloc((size_t)HID_F * OUT_F * 2));
    _Float16* fW2r   = (_Float16*)(ws + alloc((size_t)HID_F * OUT_F * 2));

    // ---- zero accumulators (fresh every launch; graph-capture safe) ----
    hipMemsetAsync(agg1, 0, (size_t)N * IN_F  * 2, stream);
    hipMemsetAsync(agg2, 0, (size_t)N * HID_F * 2, stream);
    hipMemsetAsync(cnt,  0, (size_t)N * 4,         stream);

    // ---- pack weights into WMMA B-fragments, cast x to f16 ----
    {
        int t1 = IN_F * HID_F, t2 = HID_F * OUT_F;
        pack_B_frag<<<(t1 + 255) / 256, 256, 0, stream>>>(W1l, fW1l, IN_F,  HID_F);
        pack_B_frag<<<(t1 + 255) / 256, 256, 0, stream>>>(W1r, fW1r, IN_F,  HID_F);
        pack_B_frag<<<(t2 + 255) / 256, 256, 0, stream>>>(W2l, fW2l, HID_F, OUT_F);
        pack_B_frag<<<(t2 + 255) / 256, 256, 0, stream>>>(W2r, fW2r, HID_F, OUT_F);
        int tx = N * IN_F / 4;
        cast_f32_to_f16x4<<<(tx + 255) / 256, 256, 0, stream>>>(x, x16, tx);
    }

    // ---- layer 1: fused degree + scatter (packed f16 atomics) ----
    scatter_layer1<<<(E * 32 + 255) / 256, 256, 0, stream>>>(ei, x16, agg1, cnt, E);
    {
        int t = N * IN_F / 8;   // log2(128) = 7
        normalize_f16x8<<<(t + 255) / 256, 256, 0, stream>>>(agg1, cnt, a1_16, t, 7);
    }

    // ---- layer 1 fused dual GEMM: h = relu(a1 @ W1l + x @ W1r + b1) ----
    {
        int mgroups = N / 32, ngroups = HID_F / 64;
        int waves = mgroups * ngroups;
        gemm_dual_wmma<true, true, IN_F, HID_F / 16>
            <<<(waves * 32 + 127) / 128, 128, 0, stream>>>(
                a1_16, fW1l, x16, fW1r, b1, h16, nullptr, mgroups, ngroups);
    }

    // ---- layer 2: scatter mean-aggregate over h (packed f16 atomics) ----
    scatter_layer2<<<(E * 32 + 255) / 256, 256, 0, stream>>>(ei, h16, agg2, E);
    {
        int t = N * HID_F / 8;  // log2(512) = 9
        normalize_f16x8<<<(t + 255) / 256, 256, 0, stream>>>(agg2, cnt, a2_16, t, 9);
    }

    // ---- layer 2 fused dual GEMM: out = a2 @ W2l + h @ W2r + b2 ----
    {
        int mgroups = N / 32, ngroups = OUT_F / 64;
        int waves = mgroups * ngroups;
        gemm_dual_wmma<false, false, HID_F, OUT_F / 16>
            <<<(waves * 32 + 127) / 128, 128, 0, stream>>>(
                a2_16, fW2l, h16, fW2r, b2, nullptr, out, mgroups, ngroups);
    }
}
